// VectorQuantizer_68289980007212
// MI455X (gfx1250) — matrix operand order repaired and checked by reading, not compiled, not run on hardware
//
#include <hip/hip_runtime.h>
#include <stdint.h>

// VQ-VAE vector quantizer for gfx1250 (MI455X).
//   z:        (16, 64, 64, 64) f32  -> N=65536 rows of D=64
//   codebook: (1024, 64) f32
// d_out (float): [0,4194304) quantized (B,C,H,W) | [4194304] loss | [4194305,..) indices
//
// Distances via v_wmma_f32_16x16x32_f16 (f16 in / f32 accumulate: codebook ~1e-3,
// z ~ N(0,1) -> f16 quantization error ~4e-6 in scores vs ~1e-3 inter-code gaps).
// Codebook staged into LDS via global_load_async_to_lds_b128 (ASYNCcnt path),
// shared by the 8 waves of a block: cuts codebook L2 traffic 8x.

#define K_CODES    1024
#define DIM        64
#define HW         4096          // 64*64
#define N_TOT      65536
#define CHW        262144        // 64*4096
#define Q_ELEMS    4194304
#define WAVES_PB   8
#define ROWS_PW    32            // two 16-row WMMA tiles per wave
#define ROWS_PB    (WAVES_PB * ROWS_PW)   // 256
#define GRID_MAIN  (N_TOT / ROWS_PB)      // 256
#define KCHUNK     512                    // codes staged per LDS pass
#define CHUNK_H    (KCHUNK * DIM)         // halves per chunk (64KB)

typedef __attribute__((ext_vector_type(16))) _Float16 v16h;
typedef __attribute__((ext_vector_type(8)))  float    v8f;

// ---- CDNA5 async global->LDS copy (ASYNCcnt tracked) ----
__device__ __forceinline__ void async_ld16(uint32_t lds_off, const void* g) {
  asm volatile("global_load_async_to_lds_b128 %0, %1, off"
               :: "v"(lds_off), "v"(g) : "memory");
}
__device__ __forceinline__ void wait_async0() {
  asm volatile("s_wait_asynccnt 0x0" ::: "memory");
}

// ---------------- prep: codebook f32 -> f16 + squared norms ----------------
__global__ __launch_bounds__(32) void vq_prep(const float* __restrict__ cb,
                                              _Float16* __restrict__ cbh,
                                              float* __restrict__ enorm) {
  const int k = blockIdx.x;
  const int lane = threadIdx.x;
  float s = 0.f;
#pragma unroll
  for (int j = 0; j < 2; ++j) {
    const int d = lane + j * 32;
    const float v = cb[k * DIM + d];
    cbh[k * DIM + d] = (_Float16)v;
    s += v * v;
  }
#pragma unroll
  for (int m = 16; m >= 1; m >>= 1) s += __shfl_xor(s, m, 32);
  if (lane == 0) enorm[k] = s;
}

// ---------------- main: LDS-staged WMMA distance sweep + argmin + gather ----
__global__ __launch_bounds__(256) void vq_main(const float* __restrict__ z,
                                               const float* __restrict__ cb,
                                               const _Float16* __restrict__ cbh,
                                               const float* __restrict__ enorm,
                                               float* __restrict__ qout,
                                               float* __restrict__ idxout,
                                               float* __restrict__ partial) {
  __shared__ _Float16 smem[CHUNK_H];   // 64KB codebook chunk; reused for idx/loss

  const int tid  = threadIdx.x;
  const int wave = tid >> 5;
  const int lane = tid & 31;
  const int r    = lane & 15;          // tile row / B column
  const int hs   = lane >> 4;          // half-wave select
  const int n0   = blockIdx.x * ROWS_PB + wave * ROWS_PW;

  const uint32_t lds_base = (uint32_t)(uintptr_t)&smem[0];

  // ---- issue async stage of chunk 0 (overlaps with A-tile loads) ----
  {
    const char* src = (const char*)cbh + (size_t)tid * 256;
    const uint32_t dst = lds_base + tid * 256;
#pragma unroll
    for (int i = 0; i < 16; ++i) async_ld16(dst + i * 16, src + i * 16);
  }

  // ---- load two A tiles (rows n0..n0+31), f32 -> f16, contiguous k order (same as B) ----
  // lanes 0-15 row M=lane: halves 0..15 = K 0..15;
  // lanes 16-31 row M=lane-16: K 16..31.
  v16h a0[2], a1[2];
  int  bbv[2], hwv[2];
  const float* zp[2];
#pragma unroll
  for (int t = 0; t < 2; ++t) {
    const int nn = n0 + t * 16 + r;
    bbv[t] = nn >> 12;
    hwv[t] = nn & (HW - 1);
    zp[t]  = z + (size_t)bbv[t] * CHW + hwv[t];
#pragma unroll
    for (int j = 0; j < 8; ++j) {
      const int dA = hs * 16 + j;
      a0[t][j]     = (_Float16)zp[t][(dA     ) * HW];
      a0[t][j + 8] = (_Float16)zp[t][(dA + 8) * HW];
      a1[t][j]     = (_Float16)zp[t][(dA + 32) * HW];
      a1[t][j + 8] = (_Float16)zp[t][(dA + 40) * HW];
    }
  }

  float minv[2][8];
  int   mini[2][8];
#pragma unroll
  for (int t = 0; t < 2; ++t)
#pragma unroll
    for (int v = 0; v < 8; ++v) { minv[t][v] = 3.4e38f; mini[t][v] = 0; }

  // ---- two staged passes over the codebook ----
  for (int chunk = 0; chunk < 2; ++chunk) {
    if (chunk) {
      __syncthreads();                       // all waves done reading prev chunk
      const char* src = (const char*)cbh + (size_t)CHUNK_H * 2 + (size_t)tid * 256;
      const uint32_t dst = lds_base + tid * 256;
#pragma unroll
      for (int i = 0; i < 16; ++i) async_ld16(dst + i * 16, src + i * 16);
    }
    wait_async0();
    __syncthreads();                         // chunk visible to all waves

    // B (32x16 f16), contiguous k as in A: column N = lane%16; lanes 0-15 halves = K 0..15,
    // lanes 16-31 halves = K 16..31 -> two 32B ds reads per k-tile.
    const _Float16* bp = smem + r * DIM + hs * 16;
    const float* enp = enorm + chunk * KCHUNK + r;
    for (int kt = 0; kt < KCHUNK / 16; ++kt) {
      const v16h b0 = *(const v16h*)(bp);
      const v16h b1 = *(const v16h*)(bp + 32);
      bp += 16 * DIM;
      const int   kcol = chunk * KCHUNK + kt * 16 + r;
      const float en   = enp[kt * 16];
#pragma unroll
      for (int t = 0; t < 2; ++t) {
        v8f acc = {};
        acc = __builtin_amdgcn_wmma_f32_16x16x32_f16(false, a0[t], false, b0,
                                                     (short)0, acc, false, false);
        acc = __builtin_amdgcn_wmma_f32_16x16x32_f16(false, a1[t], false, b1,
                                                     (short)0, acc, false, false);
        // C layout: VGPR v -> row M = v + 8*hs; column = kcol
#pragma unroll
        for (int v = 0; v < 8; ++v) {
          const float d = __builtin_fmaf(-2.f, acc[v], en);
          if (d < minv[t][v]) { minv[t][v] = d; mini[t][v] = kcol; }
        }
      }
    }
  }

  // ---- argmin across the 16 lanes of each half ----
#pragma unroll
  for (int t = 0; t < 2; ++t)
#pragma unroll
    for (int v = 0; v < 8; ++v)
#pragma unroll
      for (int m = 1; m < 16; m <<= 1) {
        const float ov = __shfl_xor(minv[t][v], m, 32);
        const int   oi = __shfl_xor(mini[t][v], m, 32);
        if (ov < minv[t][v] || (ov == minv[t][v] && oi < mini[t][v])) {
          minv[t][v] = ov; mini[t][v] = oi;
        }
      }

  __syncthreads();                           // smem codebook no longer needed
  int*   s_idx = (int*)smem;                 // [WAVES_PB][32]
  float* s_ls  = (float*)(smem + 2 * WAVES_PB * ROWS_PW); // past idx region

  if (lane == 0) {
#pragma unroll
    for (int v = 0; v < 8; ++v) {
      s_idx[wave * 32 + v]      = mini[0][v];
      s_idx[wave * 32 + 16 + v] = mini[1][v];
    }
  }
  if (lane == 16) {
#pragma unroll
    for (int v = 0; v < 8; ++v) {
      s_idx[wave * 32 + 8 + v]  = mini[0][v];
      s_idx[wave * 32 + 24 + v] = mini[1][v];
    }
  }
  __syncthreads();

  // ---- gather quantized rows (f32 codebook), write output + loss ----
  float ls = 0.f;
#pragma unroll
  for (int t = 0; t < 2; ++t) {
    const int myidx = s_idx[wave * 32 + t * 16 + r];
    if (hs == 0) idxout[n0 + t * 16 + r] = (float)myidx;
    const float* cq = cb + (size_t)myidx * DIM;
    float* op = qout + (size_t)bbv[t] * CHW + hwv[t];
#pragma unroll 8
    for (int d0 = 0; d0 < DIM; d0 += 2) {
      const int d = d0 + hs;                 // half-waves split even/odd d planes
      const float qv = cq[d];
      const float zv = zp[t][d * HW];
      op[d * HW] = qv;                       // 16 consecutive addrs per half-wave
      const float df = qv - zv;
      ls += df * df;
    }
  }
#pragma unroll
  for (int m = 16; m >= 1; m >>= 1) ls += __shfl_xor(ls, m, 32);
  if (lane == 0) s_ls[wave] = ls;
  __syncthreads();
  if (tid == 0) {
    float bs = 0.f;
#pragma unroll
    for (int w = 0; w < WAVES_PB; ++w) bs += s_ls[w];
    partial[blockIdx.x] = bs;
  }
}

// ---------------- deterministic loss finalize ----------------
__global__ __launch_bounds__(32) void vq_loss_finalize(const float* __restrict__ partial,
                                                       float* __restrict__ loss) {
  const int lane = threadIdx.x;
  float s = 0.f;
  for (int i = lane; i < GRID_MAIN; i += 32) s += partial[i];
#pragma unroll
  for (int m = 16; m >= 1; m >>= 1) s += __shfl_xor(s, m, 32);
  if (lane == 0) *loss = s * (1.25f / (float)Q_ELEMS);
}

extern "C" void kernel_launch(void* const* d_in, const int* in_sizes, int n_in,
                              void* d_out, int out_size, void* d_ws, size_t ws_size,
                              hipStream_t stream) {
  const float* z  = (const float*)d_in[0];
  const float* cb = (const float*)d_in[1];

  float* out   = (float*)d_out;
  float* qout  = out;
  float* lossp = out + Q_ELEMS;
  float* idxp  = out + Q_ELEMS + 1;

  // workspace: f16 codebook (128KB) | norms (4KB) | per-block loss partials (1KB)
  _Float16* cbh   = (_Float16*)d_ws;
  float*    enorm = (float*)((char*)d_ws + (size_t)K_CODES * DIM * sizeof(_Float16));
  float*    part  = enorm + K_CODES;

  vq_prep<<<K_CODES, 32, 0, stream>>>(cb, cbh, enorm);
  vq_main<<<GRID_MAIN, 256, 0, stream>>>(z, cb, cbh, enorm, qout, idxp, part);
  vq_loss_finalize<<<1, 32, 0, stream>>>(part, lossp);
}
